// RoPE_Attn_Orca_15144054686379
// MI455X (gfx1250) — matrix operand order repaired and checked by reading, not compiled, not run on hardware
//
#include <hip/hip_runtime.h>

// ---------------------------------------------------------------------------
// RoPE(dup-half variant) + softmax attention for MI455X (gfx1250, wave32).
// B=4, S=2048, H=16, D=128. Flash-attention structure:
//   - QK^T via f16x3 split (Qhi*Khi + Qlo*Khi + Qhi*Klo) -> ~2^-21 rel accuracy
//   - scores kept in log2 domain: softmax uses bare v_exp_f32 (exp2)
//   - softmax denominator accumulated as a WMMA tile against an all-ones B
//   - keys processed high->low; accumulator rescale skipped behind a
//     wave-uniform branch once the running max settles
//   - B-fragment loads software-pipelined one step ahead of the WMMAs
// All matrix math on v_wmma_f32_16x16x32_f16. 64 keys per iteration.
// ---------------------------------------------------------------------------

typedef __attribute__((ext_vector_type(16))) _Float16 v16h;
typedef __attribute__((ext_vector_type(8)))  _Float16 v8h;
typedef __attribute__((ext_vector_type(2)))  _Float16 v2h;
typedef __attribute__((ext_vector_type(8)))  float    v8f;

#define Bn 4
#define Sn 2048
#define Hn 16
#define Dn 128
#define HALFn 64
#define KT 64                 // keys per iteration
#define WAVES 8               // waves per workgroup -> 128 queries/WG
#define ROWSTRIDE (Hn * Dn)   // stride between sequence positions (floats)

// v_cvt_pk_rtz_f16_f32: pack two f32 -> two f16 in one VALU op.
__device__ __forceinline__ v2h pk2h(float a, float b) {
  return __builtin_bit_cast(v2h, __builtin_amdgcn_cvt_pkrtz(a, b));
}

// 10000^(-i/64) = exp(-i * ln(10000)/64)
__device__ __forceinline__ float inv_freq(int i) {
  return __expf(-0.14391156831212787f * (float)i);
}

__global__ __launch_bounds__(256)
void rope_attn_fa(const float* __restrict__ gq, const float* __restrict__ gk,
                  const float* __restrict__ gv, float* __restrict__ gout) {
  // LDS tiles (48 KB). K stored [key][dim] (B-frag = two 16B reads per lane),
  // V stored transposed [dim][key], P bounce buffer per wave.
  __shared__ __align__(32) _Float16 sKhi[KT][HALFn];   // 8 KB
  __shared__ __align__(32) _Float16 sKlo[KT][HALFn];   // 8 KB
  __shared__ __align__(32) _Float16 sV[Dn][KT];        // 16 KB
  __shared__ __align__(32) _Float16 sP[WAVES][16][KT]; // 16 KB

  const int tid  = threadIdx.x;
  const int wave = tid >> 5;
  const int lane = tid & 31;
  const int nl   = lane & 15;   // column / M index within halves
  const int hi   = lane >> 4;   // 0 = lanes 0-15, 1 = lanes 16-31

  const int bid = blockIdx.x;
  const int qb  = bid & 15;          // fastest: 16 q-blocks share (b,h) in L2
  const int h   = (bid >> 4) & 15;
  const int b   = bid >> 8;

  const int q0 = qb * 128 + wave * 16;     // first query row of this wave

  // ---- Q fragments (A-layout, f16 hi/lo split, live in VGPRs) ----
  const int sq = q0 + nl;
  // folds (2*sq), the 1/D*2 factor, and 1/ln2 (log2-domain softmax)
  const float qscale = 0.03125f * 1.4426950408889634f * (float)sq;
  const float* qrow = gq + ((size_t)(b * Sn + sq) * Hn + h) * Dn;
  v16h aQhi[2], aQlo[2];
#pragma unroll
  for (int f = 0; f < 2; ++f) {
    const int b0 = f * 32 + 8 * hi;
    const int b1 = f * 32 + 16 + 8 * hi;
#pragma unroll
    for (int j = 0; j < 8; ++j) {
      float x0 = qrow[b0 + j] * qscale * inv_freq(b0 + j);
      float x1 = qrow[b1 + j] * qscale * inv_freq(b1 + j);
      v2h hh = pk2h(x0, x1);
      v2h ll = pk2h(x0 - (float)hh.x, x1 - (float)hh.y);
      aQhi[f][j] = hh.x; aQhi[f][8 + j] = hh.y;
      aQlo[f][j] = ll.x; aQlo[f][8 + j] = ll.y;
    }
  }

  // all-ones B matrix (layout-invariant) for row-sum accumulation
  v16h ones;
#pragma unroll
  for (int j = 0; j < 16; ++j) ones[j] = (_Float16)1.0f;

  // ---- accumulators & online-softmax state ----
  v8f O[8];
  v8f Osum = (v8f){0.f,0.f,0.f,0.f,0.f,0.f,0.f,0.f};   // row-sums (softmax denom)
  float mrow[8];
#pragma unroll
  for (int j = 0; j < 8; ++j) O[j] = (v8f){0.f,0.f,0.f,0.f,0.f,0.f,0.f,0.f};
#pragma unroll
  for (int r = 0; r < 8; ++r) mrow[r] = -3.0e38f;

  // cooperative tile-load roles (256 threads, 64 keys)
  const int lkey = tid >> 2;            // K: key 0..63 per thread
  const int kd0  = (tid & 3) * 16;      // K: 16 dims per thread
  const int vdim = tid & 127;           // V: one dim-column per thread
  const int vkg  = (tid >> 7) * 32;     // V: key group (0 or 32)
  float invK[16];
#pragma unroll
  for (int j = 0; j < 16; ++j) invK[j] = inv_freq(kd0 + j);

  const float* kbase = gk + ((size_t)(b * Sn) * Hn + h) * Dn;
  const float* vbase = gv + ((size_t)(b * Sn) * Hn + h) * Dn;

  // B-fragment loaders (two 16B reads per lane, WMMA B table k order)
#define LOADB(T, F, BH, BL)                                                \
  do {                                                                     \
    const _Float16* ph_ = &sKhi[(T) * 16 + nl][(F) * 32 + 8 * hi];         \
    const _Float16* pl_ = &sKlo[(T) * 16 + nl][(F) * 32 + 8 * hi];         \
    ((v8h*)&(BH))[0] = *(const v8h*)(ph_);                                 \
    ((v8h*)&(BH))[1] = *(const v8h*)(ph_ + 16);                            \
    ((v8h*)&(BL))[0] = *(const v8h*)(pl_);                                 \
    ((v8h*)&(BL))[1] = *(const v8h*)(pl_ + 16);                            \
  } while (0)
#define LOADV(J, G, BV)                                                    \
  do {                                                                     \
    const _Float16* pv_ = &sV[16 * (J) + nl][(G) * 32 + 8 * hi];           \
    ((v8h*)&(BV))[0] = *(const v8h*)(pv_);                                 \
    ((v8h*)&(BV))[1] = *(const v8h*)(pv_ + 16);                            \
  } while (0)

  // keys high -> low: running max settles on the first tile (scores ~ s_k)
  for (int kb = Sn - KT; kb >= 0; kb -= KT) {
    __syncthreads();   // previous tile fully consumed
    // ---- stage K tile (RoPE-scaled, f16 hi+lo, packed converts) ----
    const int skey = kb + lkey;
    const float  fs = 2.0f * (float)skey;
    const float* krow = kbase + (size_t)skey * ROWSTRIDE + kd0;
#pragma unroll
    for (int j = 0; j < 16; j += 2) {
      float x0 = krow[j]     * (fs * invK[j]);
      float x1 = krow[j + 1] * (fs * invK[j + 1]);
      v2h hh = pk2h(x0, x1);
      v2h ll = pk2h(x0 - (float)hh.x, x1 - (float)hh.y);
      *(v2h*)&sKhi[lkey][kd0 + j] = hh;
      *(v2h*)&sKlo[lkey][kd0 + j] = ll;
    }
    // ---- stage V tile transposed: one dim-column x 32 keys per thread ----
    const float* vcol = vbase + (size_t)(kb + vkg) * ROWSTRIDE + vdim;
#pragma unroll
    for (int s = 0; s < 4; ++s) {
      v8h vb;
#pragma unroll
      for (int jj = 0; jj < 4; ++jj) {
        const int j = s * 8 + 2 * jj;
        float a0 = vcol[(size_t)j * ROWSTRIDE];
        float a1 = vcol[(size_t)(j + 1) * ROWSTRIDE];
        v2h p = pk2h(a0, a1);
        vb[2 * jj] = p.x; vb[2 * jj + 1] = p.y;
      }
      *(v8h*)&sV[vdim][vkg + s * 8] = vb;
    }
    // prefetch next (lower) tile (global_prefetch_b8)
    if (kb > 0) {
      __builtin_prefetch(krow - (size_t)KT * ROWSTRIDE, 0, 0);
      __builtin_prefetch(vcol - (size_t)KT * ROWSTRIDE, 0, 0);
    }
    __syncthreads();   // tile ready

    // ---- scores: S = Qs * Ks^T via f16x3, loads pipelined 1 step ahead ----
    v8f sc[4];
    {
      v16h bh, bl, nbh, nbl;
      LOADB(0, 0, bh, bl);
      v8f c = (v8f){0.f,0.f,0.f,0.f,0.f,0.f,0.f,0.f};
#pragma unroll
      for (int step = 0; step < 8; ++step) {
        const int t = step >> 1, f = step & 1;
        if (step < 7) LOADB((step + 1) >> 1, (step + 1) & 1, nbh, nbl);
        c = __builtin_amdgcn_wmma_f32_16x16x32_f16(false, aQhi[f], false, bh,
                                                   (short)0, c, false, false);
        c = __builtin_amdgcn_wmma_f32_16x16x32_f16(false, aQlo[f], false, bh,
                                                   (short)0, c, false, false);
        c = __builtin_amdgcn_wmma_f32_16x16x32_f16(false, aQhi[f], false, bl,
                                                   (short)0, c, false, false);
        if (f == 1) {
          sc[t] = c;
          c = (v8f){0.f,0.f,0.f,0.f,0.f,0.f,0.f,0.f};
        }
        bh = nbh; bl = nbl;
      }
    }

    // ---- online softmax: row max (row = r + 8*hi, spread over 16 lanes) ----
    float pm[8];
#pragma unroll
    for (int r = 0; r < 8; ++r)
      pm[r] = fmaxf(fmaxf(sc[0][r], sc[1][r]), fmaxf(sc[2][r], sc[3][r]));
#pragma unroll
    for (int off = 1; off < 16; off <<= 1)
#pragma unroll
      for (int r = 0; r < 8; ++r)
        pm[r] = fmaxf(pm[r], __shfl_xor(pm[r], off, 32));

    // rescale accumulators only when the running max actually moves
    int upd = 0;
#pragma unroll
    for (int r = 0; r < 8; ++r) upd |= (pm[r] > mrow[r]) ? 1 : 0;
    if (__any(upd)) {
#pragma unroll
      for (int r = 0; r < 8; ++r) {
        float mn = fmaxf(mrow[r], pm[r]);
        float al = __builtin_amdgcn_exp2f(mrow[r] - mn);   // alpha (log2 domain)
        mrow[r] = mn;
        Osum[r] *= al;
#pragma unroll
        for (int j = 0; j < 8; ++j) O[j][r] *= al;
      }
    }
    // p = exp2(score - max): bare v_exp_f32, no mul
#pragma unroll
    for (int r = 0; r < 8; ++r)
#pragma unroll
      for (int t = 0; t < 4; ++t)
        sc[t][r] = __builtin_amdgcn_exp2f(sc[t][r] - mrow[r]);

    // ---- C-layout P -> LDS -> A-layout (packed converts) ----
#pragma unroll
    for (int t = 0; t < 4; t += 2)
#pragma unroll
      for (int r = 0; r < 8; ++r) {
        v2h p = pk2h(sc[t][r], sc[t + 1][r]);
        sP[wave][r + 8 * hi][t * 16 + nl]       = p.x;
        sP[wave][r + 8 * hi][(t + 1) * 16 + nl] = p.y;
      }
    asm volatile("s_wait_dscnt 0" ::: "memory");  // same-wave LDS RAW fence
    v16h aP[2];
#pragma unroll
    for (int g = 0; g < 2; ++g) {
      const _Float16* pp = &sP[wave][nl][g * 32];
      ((v8h*)&aP[g])[0] = *(const v8h*)(pp + 8 * hi);
      ((v8h*)&aP[g])[1] = *(const v8h*)(pp + 16 + 8 * hi);
    }

    // ---- PV + denominator, g-outer so consecutive WMMAs hit different O[j]
    //      and the first V fragment load hides behind the Osum WMMAs ----
    {
      v16h bv, nbv;
      LOADV(0, 0, bv);   // in flight during the two Osum WMMAs
#pragma unroll
      for (int g = 0; g < 2; ++g)
        Osum = __builtin_amdgcn_wmma_f32_16x16x32_f16(false, aP[g], false, ones,
                                                      (short)0, Osum, false, false);
#pragma unroll
      for (int step = 0; step < 16; ++step) {
        const int g = step >> 3, j = step & 7;
        if (step < 15) {
          const int ns = step + 1;
          LOADV(ns & 7, ns >> 3, nbv);
        }
        O[j] = __builtin_amdgcn_wmma_f32_16x16x32_f16(false, aP[g], false, bv,
                                                      (short)0, O[j], false, false);
        bv = nbv;
      }
    }
  }

  // ---- normalize and store (out is [B*S, H*D] == [B,S,H,D] flat) ----
  float rinv[8];
#pragma unroll
  for (int r = 0; r < 8; ++r) rinv[r] = 1.0f / Osum[r];
#pragma unroll
  for (int r = 0; r < 8; ++r) {
    const int m = r + 8 * hi;
    float* orow = gout + ((size_t)(b * Sn + q0 + m) * Hn + h) * Dn;
#pragma unroll
    for (int j = 0; j < 8; ++j)
      orow[16 * j + nl] = O[j][r] * rinv[r];
  }
}

extern "C" void kernel_launch(void* const* d_in, const int* in_sizes, int n_in,
                              void* d_out, int out_size, void* d_ws, size_t ws_size,
                              hipStream_t stream) {
  (void)in_sizes; (void)n_in; (void)d_ws; (void)ws_size; (void)out_size;
  const float* q = (const float*)d_in[0];
  const float* k = (const float*)d_in[1];
  const float* v = (const float*)d_in[2];
  float* out = (float*)d_out;
  dim3 grid(Bn * Hn * (Sn / 128));  // 1024 workgroups, qb fastest for L2 reuse
  dim3 block(256);                  // 8 wave32 waves
  rope_attn_fa<<<grid, block, 0, stream>>>(q, k, v, out);
}
